// LSTM_q_49331994362495
// MI455X (gfx1250) — 8-bit matrix operand order repaired and checked by reading, compiled, not run on hardware
//
#include <hip/hip_runtime.h>
#include <cstdint>


typedef __attribute__((ext_vector_type(8))) int v8i;

#define STEPQ    0.03125f
#define INV_STEP 32.0f
#define STEP2    0.0009765625f

__device__ __forceinline__ float fqf(float x) {
    return fminf(fmaxf(rintf(x * INV_STEP), -128.0f), 127.0f) * STEPQ;
}
__device__ __forceinline__ int q8i(float x) {
    return (int)fminf(fmaxf(rintf(x * INV_STEP), -128.0f), 127.0f);
}
__device__ __forceinline__ float sigf(float x) {
    return 1.0f / (1.0f + __expf(-x));
}

__global__ __launch_bounds__(256) void pack_weight(const float* __restrict__ W,
                                                   int* __restrict__ Wp) {
    int id   = blockIdx.x * 256 + threadIdx.x;
    int v    = id & 7;
    int lane = (id >> 3) & 31;
    int kt   = (id >> 8) & 31;
    int nt   = id >> 13;
    int n    = nt * 16 + (lane & 15);
    int k    = kt * 64 + (v >> 1) * 16 + (v & 1) * 4 + ((lane & 16) ? 8 : 0);
    const float4 w = *(const float4*)(W + (size_t)n * 2048 + k);
    Wp[id] = (q8i(w.x) & 255) | ((q8i(w.y) & 255) << 8) |
             ((q8i(w.z) & 255) << 16) | ((q8i(w.w) & 255) << 24);
}

__global__ __launch_bounds__(256) void pack_x(const float* __restrict__ X,
                                              int* __restrict__ Xp) {
    int id   = blockIdx.x * 256 + threadIdx.x;
    int v    = id & 7;
    int lane = (id >> 3) & 31;
    int kt   = (id >> 8) & 15;
    int mt   = (id >> 12) & 3;
    int s    = id >> 14;
    int m    = mt * 16 + (lane & 15);
    int k    = kt * 64 + (v >> 1) * 16 + (v & 1) * 4 + ((lane & 16) ? 8 : 0);
    const float4 x = *(const float4*)(X + (size_t)(s * 64 + m) * 1024 + k);
    Xp[id] = (q8i(x.x) & 255) | ((q8i(x.y) & 255) << 8) |
             ((q8i(x.z) & 255) << 16) | ((q8i(x.w) & 255) << 24);
}

__global__ __launch_bounds__(256) void init_state(const float* __restrict__ bias,
                                                  float* __restrict__ biasq,
                                                  int* __restrict__ hp0,
                                                  float* __restrict__ cbuf) {
    int t = blockIdx.x * 256 + threadIdx.x;
    if (t < 4096)  biasq[t] = fqf(bias[t]);
    if (t < 16384) hp0[t]   = 0;
    cbuf[t] = 0.0f;
}

__global__ __launch_bounds__(512) void lstm_step(
    const int* __restrict__ Wp, const int* __restrict__ Xp_s,
    const int* __restrict__ hIn, int8_t* __restrict__ hOut,
    const float* __restrict__ biasq, float* __restrict__ cbuf,
    float* __restrict__ outS, float* __restrict__ hTail, float* __restrict__ cTail,
    int isLast)
{
    __shared__ float ylds[4][64][16];
    const int wg   = blockIdx.x;
    const int tid  = threadIdx.x;
    const int lane = tid & 31;
    const int wave = tid >> 5;
    const int m    = wave & 3;
    const int g    = wave >> 2;
    const int nt   = g * 64 + wg;

    const int* aX = Xp_s + m * 4096 + lane * 8;
    const int* aH = hIn  + m * 4096 + lane * 8;
    const int* bW = Wp   + nt * 8192 + lane * 8;

    v8i acc = {};
    #pragma unroll
    for (int kt = 0; kt < 16; ++kt) {
        v8i a = *(const v8i*)(aX + kt * 256);
        v8i b = *(const v8i*)(bW + kt * 256);
        acc = __builtin_amdgcn_wmma_i32_16x16x64_iu8(
             true, a,  true, b, acc, false, false);
    }
    #pragma unroll
    for (int kt = 0; kt < 16; ++kt) {
        v8i a = *(const v8i*)(aH + kt * 256);
        v8i b = *(const v8i*)(bW + (16 + kt) * 256);
        acc = __builtin_amdgcn_wmma_i32_16x16x64_iu8(
             true, a,  true, b, acc, false, false);
    }

    const int col  = lane & 15;
    const int rowB = m * 16 + ((lane & 16) ? 8 : 0);
    const float bq = biasq[g * 1024 + wg * 16 + col];
    #pragma unroll
    for (int r = 0; r < 8; ++r)
        ylds[g][rowB + r][col] = (float)acc[r] * STEP2 + bq;

    __syncthreads();

    for (int e = tid; e < 1024; e += 512) {
        int b  = e >> 4;
        int jj = e & 15;
        int j  = wg * 16 + jj;
        float i_ = fqf(sigf(ylds[0][b][jj]));
        float f_ = fqf(sigf(ylds[1][b][jj]));
        float g_ = fqf(tanhf(ylds[2][b][jj]));
        float o_ = fqf(sigf(ylds[3][b][jj]));
        size_t off = (size_t)b * 1024 + j;
        float c  = cbuf[off];
        float cn = fqf(c) * f_ + i_ * g_;
        float hn = fqf(tanhf(cn)) * o_;
        cbuf[off] = cn;
        outS[off] = hn;
        int q     = q8i(hn);
        int k     = 1024 + j;
        int kt    = (k >> 6) - 16;
        int Kl    = k & 63;
        int lo    = Kl & 15;
        int lane2 = ((lo >= 8) ? 16 : 0) + (b & 15);
        int v     = (Kl >> 4) * 2 + ((lo & 7) >> 2);
        int byt   = Kl & 3;
        int mt    = b >> 4;
        hOut[(((mt * 16 + kt) * 32 + lane2) * 8 + v) * 4 + byt] = (int8_t)q;
        if (isLast) { hTail[off] = hn; cTail[off] = cn; }
    }
}

extern "C" void kernel_launch(void* const* d_in, const int* in_sizes, int n_in,
                              void* d_out, int out_size, void* d_ws, size_t ws_size,
                              hipStream_t stream) {
    const float* X    = (const float*)d_in[0];
    const float* W    = (const float*)d_in[1];
    const float* bias = (const float*)d_in[2];
    float* out = (float*)d_out;
    char*  ws  = (char*)d_ws;

    int*    Wp   = (int*)(ws);
    int*    Xp   = (int*)(ws + (size_t)(8u << 20));
    float*  bq   = (float*)(ws + (size_t)(24u << 20));
    int8_t* hp0  = (int8_t*)(ws + (size_t)(24u << 20) + 16384);
    int8_t* hp1  = hp0 + 65536;
    float*  cbuf = (float*)(ws + (size_t)(24u << 20) + 16384 + 131072);

    pack_weight<<<8192, 256, 0, stream>>>(W, Wp);
    pack_x<<<16384, 256, 0, stream>>>(X, Xp);
    init_state<<<256, 256, 0, stream>>>(bias, bq, (int*)hp0, cbuf);

    float* hT = out + (size_t)256 * 64 * 1024;
    float* cT = hT + (size_t)64 * 1024;
    for (int s = 0; s < 256; ++s) {
        const int8_t* hin  = (s & 1) ? hp1 : hp0;
        int8_t*       hout = (s & 1) ? hp0 : hp1;
        lstm_step<<<64, 512, 0, stream>>>(
            Wp, Xp + (size_t)s * 16384,
            (const int*)hin, hout, bq, cbuf,
            out + (size_t)s * 65536, hT, cT, (s == 255) ? 1 : 0);
    }
}
